// MultiLatentAttention_10350871183693
// MI455X (gfx1250) — compile-verified, matrix operand order checked by reading, not run on hardware
//
#include <hip/hip_runtime.h>
#include <hip/hip_bf16.h>

// ---------------- problem constants ----------------
#define B_    2
#define S_    2048
#define DIM_  2048
#define H_    16
#define NOPE_ 128
#define ROPE_ 64
#define VD_   128
#define QKD_  192          // NOPE+ROPE
#define KVR_  512
#define SCALE_ 0.07216878364870322f   // 1/sqrt(192)

typedef __bf16 bf16;
typedef __attribute__((ext_vector_type(16))) __bf16 v16bf;
typedef __attribute__((ext_vector_type(8)))  __bf16 v8bf;
typedef __attribute__((ext_vector_type(8)))  float  v8f;
typedef __attribute__((ext_vector_type(4)))  unsigned int u32x4;
typedef __attribute__((ext_vector_type(8)))  int i32x8;
typedef __attribute__((ext_vector_type(4)))  int i32x4;

#if defined(__has_builtin)
#if __has_builtin(__builtin_amdgcn_tensor_load_to_lds)
#define HAVE_TDM 1
#endif
#if __has_builtin(__builtin_amdgcn_s_wait_tensorcnt)
#define HAVE_TENSORCNT 1
#endif
#endif

__device__ __forceinline__ bf16 f2bf(float x) { return (bf16)x; }

__device__ __forceinline__ v8f v8f_zero() {
  v8f z;
#pragma unroll
  for (int i = 0; i < 8; ++i) z[i] = 0.0f;
  return z;
}

// Load a 16x32 bf16 A/B fragment (lane = row/col index, K contiguous in memory)
// per CDNA5 ISA 7.12.2: lanes 0-15 hold M=0..15; VGPR0..3 = K(hi*8 .. hi*8+7),
// VGPR4..7 = K(16+hi*8 .. 23+hi*8).  -> two 16-byte loads per lane.
__device__ __forceinline__ v16bf load_frag(const bf16* base, int lo, int hi, int stride) {
  const bf16* p = base + (size_t)lo * (size_t)stride + hi * 8;
  v8bf a = *(const v8bf*)(p);
  v8bf b = *(const v8bf*)(p + 16);
  return __builtin_shufflevector(a, b, 0,1,2,3,4,5,6,7,8,9,10,11,12,13,14,15);
}

__device__ __forceinline__ v8f wmma_bf16(v16bf a, v16bf b, v8f c) {
  return __builtin_amdgcn_wmma_f32_16x16x32_bf16(false, a, false, b, (short)0, c, false, false);
}

__device__ __forceinline__ float rowmax16(float v) {
  v = fmaxf(v, __shfl_xor(v, 1, 32));
  v = fmaxf(v, __shfl_xor(v, 2, 32));
  v = fmaxf(v, __shfl_xor(v, 4, 32));
  v = fmaxf(v, __shfl_xor(v, 8, 32));
  return v;
}
__device__ __forceinline__ float rowsum16(float v) {
  v += __shfl_xor(v, 1, 32);
  v += __shfl_xor(v, 2, 32);
  v += __shfl_xor(v, 4, 32);
  v += __shfl_xor(v, 8, 32);
  return v;
}

#ifdef HAVE_TDM
// Issue a TDM 2-D tile load: tile_rows x tile_k bf16 elements from a row-major
// matrix (row stride = row_stride_elems) into LDS at lds_off, padding each
// 64B (16-DWORD) row chunk with 4 DWORDs so LDS row stride = 40 elements (80B).
// 6-arg builtin form (clang-23 / therock-10.0 headers):
//   (uint32x4 g0, int32x8 g1, int32x4 g2, int32x4 g3, int32x8 pad, i32 cpol)
__device__ __forceinline__ void tdm_load_2d(unsigned lds_off, const bf16* gaddr,
                                            unsigned tile_k, unsigned tile_rows,
                                            unsigned tensor_k, unsigned tensor_rows,
                                            unsigned row_stride_elems) {
  unsigned long long ga = (unsigned long long)(uintptr_t)gaddr;
  u32x4 g0;
  g0[0] = 1u;                                   // count=1, user descriptor
  g0[1] = lds_off;                              // lds_addr (bytes)
  g0[2] = (unsigned)ga;                         // global_addr[31:0]
  g0[3] = (unsigned)(ga >> 32) | (2u << 30);    // global_addr[56:32] | type=2 (image)
  i32x8 g1;
  // data_size=1 (2 bytes) @ [17:16]; pad_enable @ [20]; pad_interval=3 (16 DW)
  // @ [24:22]; pad_amount=3 (4 DW) @ [31:25]
  g1[0] = (int)((1u << 16) | (1u << 20) | (3u << 22) | (3u << 25));
  g1[1] = (int)((tensor_k & 0xFFFFu) << 16);                         // tensor_dim0[15:0]
  g1[2] = (int)((tensor_k >> 16) | ((tensor_rows & 0xFFFFu) << 16)); // dim0 hi | dim1 lo
  g1[3] = (int)((tensor_rows >> 16) | (tile_k << 16));               // dim1 hi | tile_dim0
  g1[4] = (int)(tile_rows);                                          // tile_dim1, tile_dim2=0
  g1[5] = (int)(row_stride_elems);                                   // tensor_dim0_stride lo
  g1[6] = 0;
  g1[7] = 0;
  i32x4 z4;
  z4[0] = 0; z4[1] = 0; z4[2] = 0; z4[3] = 0;
  i32x8 z8;
#pragma unroll
  for (int i = 0; i < 8; ++i) z8[i] = 0;
  __builtin_amdgcn_tensor_load_to_lds(g0, g1, z4, z4, z8, 0);
}
__device__ __forceinline__ void tdm_wait0() {
#ifdef HAVE_TENSORCNT
  __builtin_amdgcn_s_wait_tensorcnt((short)0);
#else
  asm volatile("s_wait_tensorcnt 0x0" ::: "memory");
#endif
}
#endif

// ---------------- elementwise f32 -> bf16 cast ----------------
__global__ __launch_bounds__(256) void cast_bf16_kernel(const float* __restrict__ in,
                                                        bf16* __restrict__ out, size_t n) {
  size_t i = (size_t)blockIdx.x * 256 + threadIdx.x;
  if (i < n) out[i] = f2bf(in[i]);
}

// ---------------- generic bf16 GEMM:  C[M,N] = A[M,K] * B[N,K]^T ----------------
// block: 256 threads (8 waves). Tile 128x128.  Wave w -> rows (w/2)*32, cols (w&1)*64.
// Double-buffered LDS staging via the Tensor Data Mover (wave 0 issues DMA,
// S_WAIT_TENSORCNT + workgroup barrier publish the tile).
#define BM 128
#define BN 128
#define BK 32
#define LDK 40   // padded LDS K stride (80B rows -> 16B aligned sublanes, bank-spread)

__global__ __launch_bounds__(256) void gemm_bf16_nt(const bf16* __restrict__ A,
                                                    const bf16* __restrict__ Bw,
                                                    float* __restrict__ C,
                                                    int M, int N, int K) {
  __shared__ __align__(16) bf16 As[2][BM * LDK];
  __shared__ __align__(16) bf16 Bs[2][BN * LDK];

  const int nbase = blockIdx.x * BN;
  const int mbase = blockIdx.y * BM;
  const int t    = threadIdx.x;
  const int wave = t >> 5;
  const int lane = t & 31;
  const int lo   = lane & 15;
  const int hi   = lane >> 4;
  const int mw   = (wave >> 1) * 32;
  const int nw   = (wave & 1) * 64;

  v8f acc[2][4];
#pragma unroll
  for (int i = 0; i < 2; ++i)
#pragma unroll
    for (int j = 0; j < 4; ++j) acc[i][j] = v8f_zero();

  const unsigned nrows = (unsigned)(N - nbase);   // >=1; TDM OOB rows read as zero

#ifdef HAVE_TDM
  int buf = 0;
  if (wave == 0) {
    tdm_load_2d((unsigned)(uintptr_t)&As[0][0], A + (size_t)mbase * K,
                BK, BM, (unsigned)K, (unsigned)(M - mbase), (unsigned)K);
    tdm_load_2d((unsigned)(uintptr_t)&Bs[0][0], Bw + (size_t)nbase * K,
                BK, BN, (unsigned)K, nrows, (unsigned)K);
    tdm_wait0();
  }
  __syncthreads();

  for (int k0 = 0; k0 < K; k0 += BK) {
    const int nb = buf ^ 1;
    if (k0 + BK < K && wave == 0) {
      tdm_load_2d((unsigned)(uintptr_t)&As[nb][0], A + (size_t)mbase * K + (k0 + BK),
                  BK, BM, (unsigned)K, (unsigned)(M - mbase), (unsigned)K);
      tdm_load_2d((unsigned)(uintptr_t)&Bs[nb][0], Bw + (size_t)nbase * K + (k0 + BK),
                  BK, BN, (unsigned)K, nrows, (unsigned)K);
    }

    v16bf a0 = load_frag(&As[buf][(mw +  0) * LDK], lo, hi, LDK);
    v16bf a1 = load_frag(&As[buf][(mw + 16) * LDK], lo, hi, LDK);
    v16bf bfr[4];
#pragma unroll
    for (int j = 0; j < 4; ++j) bfr[j] = load_frag(&Bs[buf][(nw + j * 16) * LDK], lo, hi, LDK);

#pragma unroll
    for (int j = 0; j < 4; ++j) {
      acc[0][j] = wmma_bf16(a0, bfr[j], acc[0][j]);
      acc[1][j] = wmma_bf16(a1, bfr[j], acc[1][j]);
    }

    if (wave == 0) tdm_wait0();   // next buffer resident before anyone touches it
    __syncthreads();
    buf = nb;
  }
#else
  // fallback: manual global->LDS staging (single buffer)
  v8bf zb;
#pragma unroll
  for (int i = 0; i < 8; ++i) zb[i] = (bf16)0.0f;
  const int lrow = t >> 1;
  const int lcol = (t & 1) * 16;
  for (int k0 = 0; k0 < K; k0 += BK) {
    {
      const bf16* ap = A + (size_t)(mbase + lrow) * (size_t)K + k0 + lcol;
      *(v8bf*)&As[0][lrow * LDK + lcol]     = *(const v8bf*)(ap);
      *(v8bf*)&As[0][lrow * LDK + lcol + 8] = *(const v8bf*)(ap + 8);
    }
    {
      const int brow = nbase + lrow;
      if (brow < N) {
        const bf16* bp = Bw + (size_t)brow * (size_t)K + k0 + lcol;
        *(v8bf*)&Bs[0][lrow * LDK + lcol]     = *(const v8bf*)(bp);
        *(v8bf*)&Bs[0][lrow * LDK + lcol + 8] = *(const v8bf*)(bp + 8);
      } else {
        *(v8bf*)&Bs[0][lrow * LDK + lcol]     = zb;
        *(v8bf*)&Bs[0][lrow * LDK + lcol + 8] = zb;
      }
    }
    __syncthreads();
    v16bf a0 = load_frag(&As[0][(mw +  0) * LDK], lo, hi, LDK);
    v16bf a1 = load_frag(&As[0][(mw + 16) * LDK], lo, hi, LDK);
    v16bf bfr[4];
#pragma unroll
    for (int j = 0; j < 4; ++j) bfr[j] = load_frag(&Bs[0][(nw + j * 16) * LDK], lo, hi, LDK);
#pragma unroll
    for (int j = 0; j < 4; ++j) {
      acc[0][j] = wmma_bf16(a0, bfr[j], acc[0][j]);
      acc[1][j] = wmma_bf16(a1, bfr[j], acc[1][j]);
    }
    __syncthreads();
  }
#endif

  // C layout per ISA: VGPR r -> (M = r + 8*hi, N = lo)
#pragma unroll
  for (int i = 0; i < 2; ++i)
#pragma unroll
    for (int j = 0; j < 4; ++j)
#pragma unroll
      for (int r = 0; r < 8; ++r) {
        int row = mbase + mw + i * 16 + hi * 8 + r;
        int col = nbase + nw + j * 16 + lo;
        if (col < N) C[(size_t)row * (size_t)N + col] = acc[i][j][r];
      }
}

// ---------------- build q_full bf16 (B,H,S,QKD) with RoPE on last 64 ----------------
__global__ __launch_bounds__(256) void build_q_kernel(const float* __restrict__ q,
                                                      const float* __restrict__ fc,
                                                      const float* __restrict__ fs,
                                                      bf16* __restrict__ qf) {
  const size_t total = (size_t)B_ * S_ * H_ * QKD_;
  size_t idx = (size_t)blockIdx.x * 256 + threadIdx.x;
  if (idx >= total) return;
  int d = (int)(idx % QKD_);
  size_t r0 = idx / QKD_;
  int h = (int)(r0 % H_);
  size_t bs = r0 / H_;
  int s = (int)(bs % S_);
  int b = (int)(bs / S_);
  const float* qp = q + bs * (size_t)(H_ * QKD_) + (size_t)h * QKD_;
  float val;
  if (d < NOPE_) {
    val = qp[d];
  } else {
    int j = (d - NOPE_) >> 1;
    float re = qp[NOPE_ + 2 * j];
    float im = qp[NOPE_ + 2 * j + 1];
    float c = fc[(size_t)s * (ROPE_ / 2) + j];
    float sn = fs[(size_t)s * (ROPE_ / 2) + j];
    val = ((d - NOPE_) & 1) ? (re * sn + im * c) : (re * c - im * sn);
  }
  qf[(((size_t)b * H_ + h) * S_ + s) * QKD_ + d] = f2bf(val);
}

// ---------------- kv post: RMSNorm(kv[:512]) -> bf16, RoPE(kv[512:576]) -> f32 ----------------
__global__ __launch_bounds__(256) void kv_post_kernel(const float* __restrict__ kv,
                                                      const float* __restrict__ w,
                                                      const float* __restrict__ fc,
                                                      const float* __restrict__ fs,
                                                      bf16* __restrict__ kvl,
                                                      float* __restrict__ kpe) {
  const int row = blockIdx.x;            // b*S + s
  const int s = row & (S_ - 1);
  const float* x = kv + (size_t)row * (KVR_ + ROPE_);
  const int t = threadIdx.x;

  float x0 = x[t], x1 = x[t + 256];
  float ss = x0 * x0 + x1 * x1;
#pragma unroll
  for (int off = 16; off > 0; off >>= 1) ss += __shfl_xor(ss, off, 32);

  __shared__ float red[8];
  if ((t & 31) == 0) red[t >> 5] = ss;
  __syncthreads();
  float tot = 0.f;
#pragma unroll
  for (int i = 0; i < 8; ++i) tot += red[i];
  float rs = rsqrtf(tot / (float)KVR_ + 1e-5f);

  kvl[(size_t)row * KVR_ + t]       = f2bf(x0 * rs * w[t]);
  kvl[(size_t)row * KVR_ + t + 256] = f2bf(x1 * rs * w[t + 256]);

  if (t < ROPE_ / 2) {
    float re = x[KVR_ + 2 * t];
    float im = x[KVR_ + 2 * t + 1];
    float c = fc[(size_t)s * (ROPE_ / 2) + t];
    float sn = fs[(size_t)s * (ROPE_ / 2) + t];
    kpe[(size_t)row * ROPE_ + 2 * t]     = re * c - im * sn;
    kpe[(size_t)row * ROPE_ + 2 * t + 1] = re * sn + im * c;
  }
}

// ---------------- scatter kv_proj -> k_full (B,H,S,192) bf16 and V^T (B,H,128,S) bf16 ----------------
__global__ __launch_bounds__(256) void scatter_kv_kernel(const float* __restrict__ kvp,
                                                         const float* __restrict__ kpe,
                                                         bf16* __restrict__ kf,
                                                         bf16* __restrict__ vt) {
  const size_t total = (size_t)B_ * S_ * H_ * (QKD_ + VD_);   // 320 per (b,s,h)
  size_t idx = (size_t)blockIdx.x * 256 + threadIdx.x;
  if (idx >= total) return;
  int d = (int)(idx % (QKD_ + VD_));
  size_t r0 = idx / (QKD_ + VD_);
  int h = (int)(r0 % H_);
  size_t bs = r0 / H_;
  int s = (int)(bs % S_);
  int b = (int)(bs / S_);
  const size_t kvrow = bs * (size_t)(H_ * (NOPE_ + VD_)) + (size_t)h * (NOPE_ + VD_);
  if (d < NOPE_) {
    kf[(((size_t)b * H_ + h) * S_ + s) * QKD_ + d] = f2bf(kvp[kvrow + d]);
  } else if (d < QKD_) {
    kf[(((size_t)b * H_ + h) * S_ + s) * QKD_ + d] = f2bf(kpe[bs * ROPE_ + (d - NOPE_)]);
  } else {
    int vd = d - QKD_;
    vt[(((size_t)b * H_ + h) * VD_ + vd) * (size_t)S_ + s] = f2bf(kvp[kvrow + NOPE_ + vd]);
  }
}

// ---------------- flash attention (causal) ----------------
// grid: (S/128, B*H); block 256 = 8 waves; each wave owns 16 query rows.
__global__ __launch_bounds__(256) void fa_kernel(const bf16* __restrict__ Qf,
                                                 const bf16* __restrict__ Kf,
                                                 const bf16* __restrict__ Vt,
                                                 bf16* __restrict__ Obf) {
  const int bh = blockIdx.y;
  const int b  = bh / H_;
  const int h  = bh % H_;
  const int qbase = blockIdx.x * 128;
  const int wave = threadIdx.x >> 5;
  const int lane = threadIdx.x & 31;
  const int lo = lane & 15;
  const int hi = lane >> 4;

  const bf16* Q = Qf + ((size_t)bh * S_ + qbase + wave * 16) * QKD_;
  const bf16* K = Kf + (size_t)bh * S_ * QKD_;
  const bf16* V = Vt + (size_t)bh * VD_ * S_;

  __shared__ __align__(16) bf16 Pl[8 * 16 * LDK];
  bf16* P = &Pl[wave * 16 * LDK];

  // preload Q fragments (16 q-rows x 192 = 6 chunks of K=32)
  v16bf qfr[6];
#pragma unroll
  for (int c = 0; c < 6; ++c) qfr[c] = load_frag(Q + c * 32, lo, hi, QKD_);

  v8f o[8];
#pragma unroll
  for (int n = 0; n < 8; ++n) o[n] = v8f_zero();
  float m[8], l[8];
#pragma unroll
  for (int r = 0; r < 8; ++r) { m[r] = -1e30f; l[r] = 0.0f; }

  const int qrow_base = qbase + wave * 16 + hi * 8;
  const int kend = qbase + 128;            // causal limit for this block

  for (int kt = 0; kt < kend; kt += 32) {
    // prefetch next K tile row for this lane into cache (global_prefetch_b8)
    if (kt + 32 < kend) {
      __builtin_prefetch(K + (size_t)(kt + 32 + lo) * QKD_, 0, 1);
      __builtin_prefetch(K + (size_t)(kt + 48 + lo) * QKD_, 0, 1);
    }
    v8f s0 = v8f_zero(), s1 = v8f_zero();
    const bf16* Kb0 = K + (size_t)kt * QKD_;
    const bf16* Kb1 = K + (size_t)(kt + 16) * QKD_;
#pragma unroll
    for (int c = 0; c < 6; ++c) {
      v16bf kf0 = load_frag(Kb0 + c * 32, lo, hi, QKD_);
      v16bf kf1 = load_frag(Kb1 + c * 32, lo, hi, QKD_);
      s0 = wmma_bf16(qfr[c], kf0, s0);
      s1 = wmma_bf16(qfr[c], kf1, s1);
    }

    // online softmax per row (C layout: reg r <-> row r+8*hi, cols across lanes)
#pragma unroll
    for (int r = 0; r < 8; ++r) {
      const int qr = qrow_base + r;
      float v0 = (kt + lo <= qr)      ? s0[r] * SCALE_ : -1e30f;
      float v1 = (kt + 16 + lo <= qr) ? s1[r] * SCALE_ : -1e30f;
      float rm = rowmax16(fmaxf(v0, v1));
      float mnew = fmaxf(m[r], rm);
      float p0 = __expf(v0 - mnew);
      float p1 = __expf(v1 - mnew);
      float rs = rowsum16(p0 + p1);
      float alpha = __expf(m[r] - mnew);
      l[r] = l[r] * alpha + rs;
      m[r] = mnew;
#pragma unroll
      for (int n = 0; n < 8; ++n) o[n][r] *= alpha;
      P[(r + hi * 8) * LDK + lo]      = f2bf(p0);
      P[(r + hi * 8) * LDK + 16 + lo] = f2bf(p1);
    }

    // re-stripe P (16x32) into an A-fragment and accumulate O += P * V
    v16bf pf = load_frag(P, lo, hi, LDK);
#pragma unroll
    for (int n = 0; n < 8; ++n) {
      v16bf vf = load_frag(V + (size_t)(n * 16) * S_ + kt, lo, hi, S_);
      o[n] = wmma_bf16(pf, vf, o[n]);
    }
  }

  // normalize and write attn output as bf16 (B,S,H*VD) for the o-proj GEMM
#pragma unroll
  for (int r = 0; r < 8; ++r) {
    const int qr = qrow_base + r;
    const float inv = 1.0f / l[r];
#pragma unroll
    for (int n = 0; n < 8; ++n) {
      Obf[((size_t)b * S_ + qr) * (size_t)(H_ * VD_) + (size_t)h * VD_ + n * 16 + lo] =
          f2bf(o[n][r] * inv);
    }
  }
}

// ---------------- host orchestration ----------------
static inline unsigned cdivu(size_t a, size_t b) { return (unsigned)((a + b - 1) / b); }

extern "C" void kernel_launch(void* const* d_in, const int* in_sizes, int n_in,
                              void* d_out, int out_size, void* d_ws, size_t ws_size,
                              hipStream_t stream) {
  (void)in_sizes; (void)n_in; (void)out_size; (void)ws_size;
  const float* x    = (const float*)d_in[0];
  const float* fc   = (const float*)d_in[1];
  const float* fs   = (const float*)d_in[2];
  const float* wq   = (const float*)d_in[3];
  const float* wkva = (const float*)d_in[4];
  const float* knw  = (const float*)d_in[5];
  const float* wkvb = (const float*)d_in[6];
  const float* wo   = (const float*)d_in[7];
  float* out = (float*)d_out;

  const size_t M = (size_t)B_ * S_;       // 4096 tokens
  char* ws = (char*)d_ws;
  size_t off = 0;
  auto alloc = [&](size_t bytes) -> void* {
    void* p = ws + off;
    off = (off + bytes + 255) & ~(size_t)255;
    return p;
  };

  bf16*  x_bf    = (bf16*) alloc(M * DIM_ * sizeof(bf16));
  bf16*  wq_bf   = (bf16*) alloc((size_t)H_ * QKD_ * DIM_ * sizeof(bf16));
  bf16*  wkva_bf = (bf16*) alloc((size_t)(KVR_ + ROPE_) * DIM_ * sizeof(bf16));
  bf16*  wkvb_bf = (bf16*) alloc((size_t)H_ * (NOPE_ + VD_) * KVR_ * sizeof(bf16));
  bf16*  wo_bf   = (bf16*) alloc((size_t)DIM_ * H_ * VD_ * sizeof(bf16));
  float* q_f32   = (float*)alloc(M * (size_t)H_ * QKD_ * sizeof(float));
  float* kv_f32  = (float*)alloc(M * (size_t)(KVR_ + ROPE_) * sizeof(float));
  bf16*  kvl_bf  = (bf16*) alloc(M * KVR_ * sizeof(bf16));
  float* kpe     = (float*)alloc(M * ROPE_ * sizeof(float));
  float* kvp     = (float*)alloc(M * (size_t)H_ * (NOPE_ + VD_) * sizeof(float));
  bf16*  q_full  = (bf16*) alloc((size_t)B_ * H_ * S_ * QKD_ * sizeof(bf16));
  bf16*  k_full  = (bf16*) alloc((size_t)B_ * H_ * S_ * QKD_ * sizeof(bf16));
  bf16*  v_t     = (bf16*) alloc((size_t)B_ * H_ * VD_ * S_ * sizeof(bf16));
  bf16*  attn_bf = (bf16*) alloc(M * (size_t)H_ * VD_ * sizeof(bf16));

  auto cast = [&](const float* in, bf16* o, size_t n) {
    cast_bf16_kernel<<<cdivu(n, 256), 256, 0, stream>>>(in, o, n);
  };
  cast(x,    x_bf,    M * DIM_);
  cast(wq,   wq_bf,   (size_t)H_ * QKD_ * DIM_);
  cast(wkva, wkva_bf, (size_t)(KVR_ + ROPE_) * DIM_);
  cast(wkvb, wkvb_bf, (size_t)H_ * (NOPE_ + VD_) * KVR_);
  cast(wo,   wo_bf,   (size_t)DIM_ * H_ * VD_);

  // q = x @ wq^T   (4096 x 3072, K=2048)
  gemm_bf16_nt<<<dim3(cdivu((size_t)H_ * QKD_, BN), cdivu(M, BM)), 256, 0, stream>>>(
      x_bf, wq_bf, q_f32, (int)M, H_ * QKD_, DIM_);
  build_q_kernel<<<cdivu((size_t)B_ * S_ * H_ * QKD_, 256), 256, 0, stream>>>(
      q_f32, fc, fs, q_full);

  // kv = x @ wkv_a^T  (4096 x 576, K=2048)
  gemm_bf16_nt<<<dim3(cdivu(KVR_ + ROPE_, BN), cdivu(M, BM)), 256, 0, stream>>>(
      x_bf, wkva_bf, kv_f32, (int)M, KVR_ + ROPE_, DIM_);
  kv_post_kernel<<<(unsigned)M, 256, 0, stream>>>(kv_f32, knw, fc, fs, kvl_bf, kpe);

  // kv_proj = kv_latent @ wkv_b^T  (4096 x 4096, K=512)
  gemm_bf16_nt<<<dim3(cdivu((size_t)H_ * (NOPE_ + VD_), BN), cdivu(M, BM)), 256, 0, stream>>>(
      kvl_bf, wkvb_bf, kvp, (int)M, H_ * (NOPE_ + VD_), KVR_);
  scatter_kv_kernel<<<cdivu((size_t)B_ * S_ * H_ * (QKD_ + VD_), 256), 256, 0, stream>>>(
      kvp, kpe, k_full, v_t);

  // flash attention
  fa_kernel<<<dim3(S_ / 128, B_ * H_), 256, 0, stream>>>(q_full, k_full, v_t, attn_bf);

  // out = attn @ wo^T  (4096 x 2048, K=2048)
  gemm_bf16_nt<<<dim3(cdivu(DIM_, BN), cdivu(M, BM)), 256, 0, stream>>>(
      attn_bf, wo_bf, out, (int)M, DIM_, H_ * VD_);
}
